// SynergyModel_9844065042683
// MI455X (gfx1250) — compile-verified, matrix operand order checked by reading, not run on hardware
//
#include <hip/hip_runtime.h>
#include <hip/hip_bf16.h>
#include <math.h>

typedef _Float16 v16h __attribute__((ext_vector_type(16)));
typedef _Float16 h8   __attribute__((ext_vector_type(8)));
typedef _Float16 h2   __attribute__((ext_vector_type(2)));
typedef float    v8f  __attribute__((ext_vector_type(8)));

#define DEV __device__ __forceinline__

constexpr int B = 32768, E = 32, H = 4, FF = 4;
constexpr int DCN = 160;   // max head width (cnt); syn = 128

__constant__ int IDX_SYN_D[10][5] = {
 {0,1,2,3,4},{1,0,2,3,4},{2,0,1,3,4},{3,0,1,2,4},{4,0,1,2,3},
 {5,6,7,8,9},{6,5,7,8,9},{7,5,6,8,9},{8,5,6,7,9},{9,5,6,7,8}};
__constant__ int IDX_CNT_D[10][6] = {
 {0,5,6,7,8,9},{1,5,6,7,8,9},{2,5,6,7,8,9},{3,5,6,7,8,9},{4,5,6,7,8,9},
 {5,0,1,2,3,4},{6,0,1,2,3,4},{7,0,1,2,3,4},{8,0,1,2,3,4},{9,0,1,2,3,4}};

// ---- WMMA fragment helpers (CDNA5 16x16x32 f16, wave32) ----
// A 16x32 f16 layout (ISA 7.12.2): lane%16 = row M; lanes<16 hold K 0-7 & 16-23,
// lanes>=16 hold K 8-15 & 24-31, packed 2 halves per VGPR.
DEV v16h load_a_frag(const _Float16* buf, int D, int kt, int lane) {
  int row   = lane & 15;
  int kbase = kt * 32 + ((lane & 16) ? 8 : 0);
  const _Float16* p = buf + row * D + kbase;
  union { v16h v; h8 h[2]; } u;
  u.h[0] = *(const h8*)(p);
  u.h[1] = *(const h8*)(p + 16);
  return u.v;
}
// B fragments are pre-packed: 16 contiguous halves per (tile,lane).
DEV v16h load_b_frag(const _Float16* pk, int tIdx, int lane) {
  const h8* p = (const h8*)(pk + ((size_t)tIdx * 32 + lane) * 16);
  union { v16h v; h8 h[2]; } u;
  u.h[0] = p[0]; u.h[1] = p[1];
  return u.v;
}
DEV v8f wmma32(v16h a, v16h b, v8f c) {
  return __builtin_amdgcn_wmma_f32_16x16x32_f16(false, a, false, b, (short)0, c, false, false);
}
// slope < 1  =>  leaky(x) == max(x, 0.1x)  (2 VALU ops)
DEV float leaky(float v) { return fmaxf(v, 0.1f * v); }

// Generic DxD dense layer over PERMUTED activation storage.
// Stored column s within each 32-block maps to logical column (s&1)*16 + (s>>1);
// the weight pack compensates (permK=1), so the GEMM is unchanged. Benefit:
// the paired N-tiles' outputs land in ADJACENT halves -> one packed f16x2
// ds_store_b32 per row instead of two b16 stores.
template<int D, bool ACT>
DEV void dense_layer(const _Float16* inBuf, _Float16* outBuf,
                     const _Float16* wPack, const float* bias, int lane) {
  constexpr int Kt = D / 32, Nt = D / 16;
  int col = lane & 15;
  int rb  = (lane & 16) ? 8 : 0;
#pragma unroll
  for (int nt = 0; nt < Nt; nt += 2) {
    float b0 = bias[nt * 16 + col];
    float b1 = bias[nt * 16 + 16 + col];
    v8f c0 = {}, c1 = {};
    const _Float16* wp0 = wPack + (size_t)nt * Kt * 512;
    const _Float16* wp1 = wp0 + (size_t)Kt * 512;
#pragma unroll
    for (int kt = 0; kt < Kt; ++kt) {
      v16h a = load_a_frag(inBuf, D, kt, lane);
      c0 = wmma32(a, load_b_frag(wp0, kt, lane), c0);
      c1 = wmma32(a, load_b_frag(wp1, kt, lane), c1);
    }
#pragma unroll
    for (int r = 0; r < 8; ++r) {
      float v0 = c0[r] + b0, v1 = c1[r] + b1;
      if (ACT) { v0 = leaky(v0); v1 = leaky(v1); }
      h2 p; p[0] = (_Float16)v0; p[1] = (_Float16)v1;
      *(h2*)(outBuf + (rb + r) * D + nt * 16 + 2 * col) = p;
    }
  }
}

// ---- Weight packing into WMMA-B fragment layout (nt-major, kt inner).
// permK=1 composes the activation-storage column permutation into K.
__global__ void pack_wmma_b(const float* __restrict__ src, _Float16* __restrict__ dst,
                            int Kdim, int Ndim, int nMats, int permK) {
  int Kt = Kdim >> 5, Nt = Ndim >> 4;
  long total = (long)nMats * Kt * Nt * 512;
  long gid = (long)blockIdx.x * blockDim.x + threadIdx.x;
  if (gid >= total) return;
  int  i    = (int)(gid & 15);
  long r    = gid >> 4;
  int  lane = (int)(r & 31); r >>= 5;
  int  tpm  = Kt * Nt;
  int  tIdx = (int)(r % tpm);
  int  m    = (int)(r / tpm);
  int  nt   = tIdx / Kt, kt = tIdx - nt * Kt;
  int  j    = (i < 8 ? i : i + 8) + ((lane & 16) ? 8 : 0);   // fragment K slot
  int  kin  = permK ? ((j & 1) * 16 + (j >> 1)) : j;         // stored -> logical
  int  K = kt * 32 + kin;
  int  N = nt * 16 + (lane & 15);
  dst[gid] = (_Float16)src[((size_t)m * Kdim + K) * Ndim + N];
}

__global__ void zero_f32(float* __restrict__ p, long n) {
  long i = (long)blockIdx.x * blockDim.x + threadIdx.x;
  if (i < n) p[i] = 0.f;
}

// ---- Stage 1: xh[b*1280 + t*256 + c] = (emb[inputs[b,t]] @ hW + hb), tokens 0..4.
// The reference reshape (B,5,256)->(B,4,10,32) is flat-index identity, so the
// output row for token t is a contiguous 256-half span at b*1280 + t*256.
__global__ void embed_kernel(const int* __restrict__ inputs, const float* __restrict__ emb,
                             const _Float16* __restrict__ hWp, const float* __restrict__ hb,
                             _Float16* __restrict__ xh) {
  __shared__ alignas(16) _Float16 sE[4][16 * 32];
  __shared__ alignas(16) _Float16 sO[4][16 * 256];
  int lane = threadIdx.x & 31, w = threadIdx.x >> 5;
  int rowBase = (blockIdx.x * 4 + w) * 16;
  _Float16* a = sE[w];
  _Float16* o = sO[w];
  for (int m = 0; m < 16; ++m) {
    int n = rowBase + m; int b = n / 5; int t = n - b * 5;
    int idx = inputs[b * 10 + t];
    a[m * 32 + lane] = (_Float16)emb[idx * E + lane];
  }
  v16h af = load_a_frag(a, 32, 0, lane);
  int col = lane & 15, rb = (lane & 16) ? 8 : 0;
#pragma unroll
  for (int nt = 0; nt < 16; nt += 2) {
    float b0 = hb[nt * 16 + col];
    float b1 = hb[nt * 16 + 16 + col];
    v8f c0 = {}, c1 = {};
    c0 = wmma32(af, load_b_frag(hWp, nt, lane), c0);
    c1 = wmma32(af, load_b_frag(hWp, nt + 1, lane), c1);
#pragma unroll
    for (int r = 0; r < 8; ++r) {
      o[(rb + r) * 256 + nt * 16 + col]       = (_Float16)(c0[r] + b0);
      o[(rb + r) * 256 + (nt + 1) * 16 + col] = (_Float16)(c1[r] + b1);
    }
  }
  // Coalesced write-out: one 16-byte store per lane per row.
  for (int m = 0; m < 16; ++m) {
    int n = rowBase + m; int b = n / 5; int t = n - b * 5;
    _Float16* dst = xh + (size_t)b * 1280 + t * 256 + lane * 8;
    *(h8*)dst = *(const h8*)&o[m * 256 + lane * 8];
  }
}

// ---- Stage 2: per-head MLP, fused xm -> 4xFF -> res+l2n -> proj -> l2n -> sum_t.
// All FF activations (incl. xm) live in the permuted column layout; sums and
// element-wise ops are permutation-invariant; projection emits true order.
template<int C>
__global__ void head_kernel(const _Float16* __restrict__ xh,
                            const _Float16* __restrict__ ffPack,
                            const float* __restrict__ ffb,
                            const _Float16* __restrict__ fPack,
                            const float* __restrict__ fb,
                            float* __restrict__ feat, int featBase0) {
  constexpr int D  = (C - 1) * E;
  constexpr int Kt = D / 32;
  __shared__ alignas(16) _Float16 sXm[4][16 * DCN];
  __shared__ alignas(16) _Float16 sA [4][16 * DCN];
  __shared__ alignas(16) _Float16 sB [4][16 * DCN];
  __shared__ float sRow[4][16];
  int lane = threadIdx.x & 31, w = threadIdx.x >> 5;
  int h = blockIdx.y;
  int rowBase = (blockIdx.x * 4 + w) * 16;
  _Float16* xm = sXm[w]; _Float16* bufA = sA[w]; _Float16* bufB = sB[w];

  constexpr size_t perLayer = (size_t)Kt * (D / 16) * 512;
  const float*     fb0 = ffb + (size_t)h * FF * D;
  const _Float16*  fp0 = ffPack + (size_t)h * FF * perLayer;
  __builtin_prefetch(fp0, 0, 3);          // global_prefetch_b8: warm weight stream

  // xm in permuted layout: logical col (c-1)*32 + lane -> stored
  // (c-1)*32 + 2*(lane&15) + (lane>>4)
  int sperm = 2 * (lane & 15) + (lane >> 4);
  for (int m = 0; m < 16; ++m) {
    int n = rowBase + m; int b = n / 10; int tt = n - b * 10;
    const int* ir = (C == 5) ? &IDX_SYN_D[tt][0] : &IDX_CNT_D[tt][0];
    const _Float16* base = xh + (size_t)b * 1280 + h * 320;  // flat (B,4,10,32)
    float x0 = (float)base[ir[0] * E + lane];
#pragma unroll
    for (int c = 1; c < C; ++c) {
      float xc = (float)base[ir[c] * E + lane];
      xm[m * D + (c - 1) * E + sperm] = (_Float16)(x0 * xc);
    }
  }
  // 4 FF layers (wave-local LDS ping-pong; DS ops are in-order per wave)
  dense_layer<D, true>(xm,   bufA, fp0 + 0 * perLayer, fb0 + 0 * D, lane);
  __builtin_prefetch(fp0 + 2 * perLayer, 0, 3);
  dense_layer<D, true>(bufA, bufB, fp0 + 1 * perLayer, fb0 + 1 * D, lane);
  dense_layer<D, true>(bufB, bufA, fp0 + 2 * perLayer, fb0 + 2 * D, lane);
  dense_layer<D, true>(bufA, bufB, fp0 + 3 * perLayer, fb0 + 3 * D, lane);
  // residual + l2norm into bufA (all 32 lanes: half-row each, combine via shfl;
  // both buffers share the permuted layout, so sums/element-ops are aligned)
  {
    int rrow = lane & 15, half = lane >> 4;
    float s = 0.f;
    for (int col = half * (D / 2); col < (half + 1) * (D / 2); ++col) {
      float v = (float)bufB[rrow * D + col] + (float)xm[rrow * D + col];
      s += v * v;
    }
    s += __shfl_xor(s, 16);
    if (lane < 16) sRow[w][rrow] = rsqrtf(fmaxf(s, 1e-12f));
  }
  for (int row = 0; row < 16; ++row)
    for (int col = lane; col < D; col += 32) {
      float v = ((float)bufB[row * D + col] + (float)xm[row * D + col]) * sRow[w][row];
      bufA[row * D + col] = (_Float16)v;
    }
  // projection D -> 32 (leaky) into f32 staging in TRUE column order
  float* st = (float*)bufB;
  {
    int col = lane & 15, rb = (lane & 16) ? 8 : 0;
    const _Float16* fpk = fPack + (size_t)h * (Kt * 2) * 512;
    v8f c0 = {}, c1 = {};
    float b0 = fb[h * E + col], b1 = fb[h * E + 16 + col];
#pragma unroll
    for (int kt = 0; kt < Kt; ++kt) {
      v16h a = load_a_frag(bufA, D, kt, lane);
      c0 = wmma32(a, load_b_frag(fpk, kt, lane), c0);
      c1 = wmma32(a, load_b_frag(fpk, Kt + kt, lane), c1);
    }
#pragma unroll
    for (int r = 0; r < 8; ++r) {
      st[(rb + r) * E + col]      = leaky(c0[r] + b0);
      st[(rb + r) * E + 16 + col] = leaky(c1[r] + b1);
    }
  }
  // final l2norm + sum over t into feat (one col per lane; no intra-op conflicts)
  {
    int rrow = lane & 15, half = lane >> 4;
    float s = 0.f;
#pragma unroll
    for (int e2 = half * 16; e2 < half * 16 + 16; ++e2) {
      float v = st[rrow * E + e2]; s += v * v;
    }
    s += __shfl_xor(s, 16);
    if (lane < 16) sRow[w][rrow] = rsqrtf(fmaxf(s, 1e-12f));
  }
  for (int row = 0; row < 16; ++row) {
    int n = rowBase + row; int b = n / 10;
    float v = st[row * E + lane] * sRow[w][row];
    atomicAdd(&feat[(size_t)b * (2 * H * E) + featBase0 + h * E + lane], v);
  }
}

// ---- Stage 3: feat -> red -> comb x4 -> fin(sigmoid) ----
__global__ void final_kernel(const float* __restrict__ feat,
                             const _Float16* __restrict__ redWp, const float* __restrict__ redb,
                             const _Float16* __restrict__ combWp, const float* __restrict__ combb,
                             const float* __restrict__ finW, const float* __restrict__ finb,
                             float* __restrict__ out) {
  __shared__ alignas(16) _Float16 sF[4][16 * 256];
  __shared__ alignas(16) _Float16 sA[4][16 * 32];
  __shared__ alignas(16) _Float16 sB[4][16 * 32];
  int lane = threadIdx.x & 31, w = threadIdx.x >> 5;
  int rowBase = (blockIdx.x * 4 + w) * 16;
  _Float16* f = sF[w]; _Float16* bufA = sA[w]; _Float16* bufB = sB[w];
  for (int row = 0; row < 16; ++row)
    for (int col = lane; col < 256; col += 32)
      f[row * 256 + col] = (_Float16)feat[(size_t)(rowBase + row) * 256 + col];
  int col = lane & 15, rb = (lane & 16) ? 8 : 0;
  // red: 256 -> 32 (paired N-tiles, shared A-frag)
  {
    v8f c0 = {}, c1 = {};
    float b0 = redb[col], b1 = redb[16 + col];
#pragma unroll
    for (int kt = 0; kt < 8; ++kt) {
      v16h a = load_a_frag(f, 256, kt, lane);
      c0 = wmma32(a, load_b_frag(redWp, kt, lane), c0);
      c1 = wmma32(a, load_b_frag(redWp, 8 + kt, lane), c1);
    }
#pragma unroll
    for (int r = 0; r < 8; ++r) {
      bufA[(rb + r) * 32 + col]      = (_Float16)leaky(c0[r] + b0);
      bufA[(rb + r) * 32 + 16 + col] = (_Float16)leaky(c1[r] + b1);
    }
  }
  // comb x4 (32x32), shared A-frag for both N-tiles
  for (int l = 0; l < 4; ++l) {
    _Float16* in  = (l & 1) ? bufB : bufA;
    _Float16* ob  = (l & 1) ? bufA : bufB;
    const _Float16* wl = combWp + (size_t)l * 2 * 512;
    v8f c0 = {}, c1 = {};
    float b0 = combb[l * 32 + col], b1 = combb[l * 32 + 16 + col];
    v16h a = load_a_frag(in, 32, 0, lane);
    c0 = wmma32(a, load_b_frag(wl, 0, lane), c0);
    c1 = wmma32(a, load_b_frag(wl, 1, lane), c1);
#pragma unroll
    for (int r = 0; r < 8; ++r) {
      ob[(rb + r) * 32 + col]      = (_Float16)leaky(c0[r] + b0);
      ob[(rb + r) * 32 + 16 + col] = (_Float16)leaky(c1[r] + b1);
    }
  }
  // fin: 32 -> 1, sigmoid (last comb output landed in bufA)
  if (lane < 16) {
    float s = finb[0];
#pragma unroll
    for (int e2 = 0; e2 < 32; ++e2) s += (float)bufA[lane * 32 + e2] * finW[e2];
    out[rowBase + lane] = 1.f / (1.f + expf(-s));
  }
}

extern "C" void kernel_launch(void* const* d_in, const int* in_sizes, int n_in,
                              void* d_out, int out_size, void* d_ws, size_t ws_size,
                              hipStream_t stream) {
  const int*   inputs  = (const int*)d_in[0];
  const float* emb     = (const float*)d_in[1];
  const float* hW      = (const float*)d_in[2];
  const float* hb      = (const float*)d_in[3];
  const float* syn_ffW = (const float*)d_in[4];
  const float* syn_ffb = (const float*)d_in[5];
  const float* syn_fW  = (const float*)d_in[6];
  const float* syn_fb  = (const float*)d_in[7];
  const float* cnt_ffW = (const float*)d_in[8];
  const float* cnt_ffb = (const float*)d_in[9];
  const float* cnt_fW  = (const float*)d_in[10];
  const float* cnt_fb  = (const float*)d_in[11];
  const float* redW    = (const float*)d_in[12];
  const float* redb    = (const float*)d_in[13];
  const float* combW   = (const float*)d_in[14];
  const float* combb   = (const float*)d_in[15];
  const float* finW    = (const float*)d_in[16];
  const float* finb    = (const float*)d_in[17];
  float* out = (float*)d_out;

  char* ws = (char*)d_ws;
  _Float16* xh   = (_Float16*)ws;                       // 41,943,040 halves
  float*    feat = (float*)(ws + 83886080);             // 8,388,608 f32
  _Float16* pk   = (_Float16*)(ws + 117440512);         // packed f16 weights
  _Float16* hWp    = pk + 0;
  _Float16* synffp = pk + 8192;
  _Float16* synfp  = pk + 270336;
  _Float16* cntffp = pk + 286720;
  _Float16* cntfp  = pk + 696320;
  _Float16* redp   = pk + 716800;
  _Float16* combp  = pk + 724992;

  auto nb = [](long n) { return (unsigned)((n + 255) / 256); };
  pack_wmma_b<<<nb(8192),   256, 0, stream>>>(hW,      hWp,    32, 256, 1, 0);
  pack_wmma_b<<<nb(262144), 256, 0, stream>>>(syn_ffW, synffp, 128, 128, 16, 1);
  pack_wmma_b<<<nb(16384),  256, 0, stream>>>(syn_fW,  synfp,  128, 32, 4, 1);
  pack_wmma_b<<<nb(409600), 256, 0, stream>>>(cnt_ffW, cntffp, 160, 160, 16, 1);
  pack_wmma_b<<<nb(20480),  256, 0, stream>>>(cnt_fW,  cntfp,  160, 32, 4, 1);
  pack_wmma_b<<<nb(8192),   256, 0, stream>>>(redW,    redp,   256, 32, 1, 0);
  pack_wmma_b<<<nb(4096),   256, 0, stream>>>(combW,   combp,  32, 32, 4, 0);
  zero_f32<<<nb(8388608), 256, 0, stream>>>(feat, 8388608L);

  embed_kernel<<<2560, 128, 0, stream>>>(inputs, emb, hWp, hb, xh);
  head_kernel<5><<<dim3(5120, 4), 128, 0, stream>>>(xh, synffp, syn_ffb, synfp, syn_fb, feat, 0);
  head_kernel<6><<<dim3(5120, 4), 128, 0, stream>>>(xh, cntffp, cnt_ffb, cntfp, cnt_fb, feat, H * E);
  final_kernel<<<512, 128, 0, stream>>>(feat, redp, redb, combp, combb, finW, finb, out);
}
